// STAUCell_24661702213810
// MI455X (gfx1250) — matrix operand order repaired and checked by reading, not compiled, not run on hardware
//
#include <hip/hip_runtime.h>
#include <hip/hip_bf16.h>

// ---------------------------------------------------------------------------
// STAU cell for MI455X (gfx1250, wave32, WMMA).
//   P0: weight fp32->bf16 conversion (workspace)
//   P1: t_next/s_next = X @ W^T + b   (v_wmma_f32_16x16x32_bf16, f32 accum)
//   P2: dot/softmax/trend/gate fusion (memory-bound streaming, 512MB)
//   P3: 6-way concat GEMM fused with final cross-gating -> d_out
// ---------------------------------------------------------------------------

#define DIMK 512
#define NTOK 16384
#define TAUK 4

typedef __attribute__((ext_vector_type(16))) __bf16  bf16x16;
typedef __attribute__((ext_vector_type(4)))  __bf16  bf16x4;
typedef __attribute__((ext_vector_type(2)))  __bf16  bf16x2;
typedef __attribute__((ext_vector_type(8)))  float   v8f;
typedef __attribute__((ext_vector_type(4)))  float   f32x4;
typedef __attribute__((ext_vector_type(4)))  unsigned int u32x4;

union BFrag { bf16x16 v; u32x4 q[2]; };
union AFragCvt { bf16x16 v; bf16x2 p[8]; };

__device__ __forceinline__ __bf16 f2bf(float f) {
  unsigned u = __builtin_bit_cast(unsigned, f);
  u += 0x7FFFu + ((u >> 16) & 1u);            // round-to-nearest-even
  unsigned short h = (unsigned short)(u >> 16);
  return __builtin_bit_cast(__bf16, h);
}

__device__ __forceinline__ bf16x2 f2bf_pk(float lo, float hi) {
#if __has_builtin(__builtin_amdgcn_cvt_pk_bf16_f32)
  return __builtin_amdgcn_cvt_pk_bf16_f32(lo, hi);
#else
  bf16x2 r; r.x = f2bf(lo); r.y = f2bf(hi); return r;
#endif
}

// Build a 16-element bf16 A fragment from 4 fp32 quads (packed convert).
__device__ __forceinline__ bf16x16 cvt16(f32x4 a0, f32x4 a1, f32x4 a2, f32x4 a3) {
  AFragCvt f;
  f.p[0] = f2bf_pk(a0.x, a0.y); f.p[1] = f2bf_pk(a0.z, a0.w);
  f.p[2] = f2bf_pk(a1.x, a1.y); f.p[3] = f2bf_pk(a1.z, a1.w);
  f.p[4] = f2bf_pk(a2.x, a2.y); f.p[5] = f2bf_pk(a2.z, a2.w);
  f.p[6] = f2bf_pk(a3.x, a3.y); f.p[7] = f2bf_pk(a3.z, a3.w);
  return f.v;
}

__device__ __forceinline__ float sigmoidf_(float x) {
  return 1.0f / (1.0f + __expf(-x));
}

#define WMMA_BF16(A, B, C) \
  __builtin_amdgcn_wmma_f32_16x16x32_bf16(false, (A), false, (B), (short)0, (C), false, false)

// ---------------------------------------------------------------------------
// P0: fp32 -> bf16 bulk convert (weights only; 4 MB total, negligible)
// ---------------------------------------------------------------------------
__global__ void k_cvt_bf16(const float* __restrict__ src,
                           __bf16* __restrict__ dst, int n4) {
  int i = blockIdx.x * blockDim.x + threadIdx.x;
  if (i < n4) {
    f32x4 v = ((const f32x4*)src)[i];
    union { bf16x4 v; bf16x2 p[2]; } b;
    b.p[0] = f2bf_pk(v.x, v.y);
    b.p[1] = f2bf_pk(v.z, v.w);
    ((bf16x4*)dst)[i] = b.v;
  }
}

// ---------------------------------------------------------------------------
// P1: next-state GEMMs.  grid=(NTOK/16, 2), block=256 (8 waves).
// Wave w covers 16 tokens x cols [w*64, w*64+64). A loaded fp32, cvt in-reg.
// All 4 B fragments are loaded into distinct registers BEFORE the 4 WMMAs so
// the compiler can clause the loads and overlap them with WMMA issue.
// ---------------------------------------------------------------------------
__global__ void __launch_bounds__(256)
k_next_gemm(const float* __restrict__ T_t, const float* __restrict__ S_t,
            const __bf16* __restrict__ Wtn, const __bf16* __restrict__ Wsn,
            const float* __restrict__ btn, const float* __restrict__ bsn,
            float* __restrict__ tnext, float* __restrict__ snext) {
  const bool sSide = (blockIdx.y != 0);
  const float*  A    = sSide ? S_t  : T_t;
  const __bf16* Wb   = sSide ? Wsn  : Wtn;
  const float*  bias = sSide ? bsn  : btn;
  float*        out  = sSide ? snext : tnext;

  const int lane = threadIdx.x & 31;
  const int wave = threadIdx.x >> 5;
  const int lm = lane & 15;
  const int lh = lane >> 4;        // half-wave select
  const int cb = lh * 8;           // A-fragment K-chunk base (ISA 16-bit A layout)
  const int m0 = blockIdx.x * 16;
  const int j0 = wave * 64;

  v8f acc[4] = {};
  const float* Arow = A + (size_t)(m0 + lm) * DIMK;
  const __bf16* Wrow = Wb + (size_t)(j0 + lm) * DIMK + lh * 8;

  for (int k0 = 0; k0 < DIMK; k0 += 32) {
    // A fragment: lanes 0-15 hold K[k0..k0+8)+K[k0+16..k0+24), lanes 16-31 +8.
    f32x4 a0 = *(const f32x4*)(Arow + k0 + cb);
    f32x4 a1 = *(const f32x4*)(Arow + k0 + cb + 4);
    f32x4 a2 = *(const f32x4*)(Arow + k0 + 16 + cb);
    f32x4 a3 = *(const f32x4*)(Arow + k0 + 16 + cb + 4);

    // B = W^T (col-major) == W row-major: lane holds K chunks [8h,8h+8)+[16+8h,24+8h) of row j.
    BFrag bf[4];
#pragma unroll
    for (int t = 0; t < 4; ++t) {
      const __bf16* bp = Wrow + (size_t)(t * 16) * DIMK + k0;
      bf[t].q[0] = *(const u32x4*)(bp);
      bf[t].q[1] = *(const u32x4*)(bp + 16);
    }

    if (k0 + 32 < DIMK) __builtin_prefetch(Arow + k0 + 32, 0, 1);

    const bf16x16 av = cvt16(a0, a1, a2, a3);

#pragma unroll
    for (int t = 0; t < 4; ++t) acc[t] = WMMA_BF16(av, bf[t].v, acc[t]);
  }

#pragma unroll
  for (int t = 0; t < 4; ++t) {
    const int col = j0 + t * 16 + lm;
    const float bv = bias[col];
#pragma unroll
    for (int r = 0; r < 8; ++r) {
      const int row = m0 + lh * 8 + r;     // ISA f32 C/D layout
      out[(size_t)row * DIMK + col] = acc[t][r] + bv;
    }
  }
}

// ---------------------------------------------------------------------------
// P2: fusion (memory-bound). One wave per token; lane owns interleaved
// d-slices (i*32+lane)*4 .. +4 so every b128 load is a contiguous 512B per wave.
// grid = NTOK/8, block = 256.
// ---------------------------------------------------------------------------
__global__ void __launch_bounds__(256)
k_fuse(const float* __restrict__ T_t, const float* __restrict__ S_t,
       const float* __restrict__ t_att, const float* __restrict__ s_att,
       const float* __restrict__ t_spatial, const float* __restrict__ s_spatial,
       const float* __restrict__ t_next, const float* __restrict__ s_next,
       __bf16* __restrict__ Tfus, __bf16* __restrict__ Sfus) {
  const int lane = threadIdx.x & 31;
  const int wave = threadIdx.x >> 5;
  const int n = blockIdx.x * 8 + wave;
  const size_t base = (size_t)n * DIMK;

  float tn[16], sn[16];
#pragma unroll
  for (int i = 0; i < 4; ++i) {
    const size_t off = base + (size_t)(i * 32 + lane) * 4;
    f32x4 v = *(const f32x4*)(t_next + off);
    f32x4 w = *(const f32x4*)(s_next + off);
    tn[i * 4 + 0] = v.x; tn[i * 4 + 1] = v.y; tn[i * 4 + 2] = v.z; tn[i * 4 + 3] = v.w;
    sn[i * 4 + 0] = w.x; sn[i * 4 + 1] = w.y; sn[i * 4 + 2] = w.z; sn[i * 4 + 3] = w.w;
  }

  // dot products over feature dim
  float dps[TAUK], dpt[TAUK];
#pragma unroll
  for (int k = 0; k < TAUK; ++k) {
    const size_t kb = (size_t)k * NTOK * DIMK + base;
    float accs = 0.f, acct = 0.f;
#pragma unroll
    for (int i = 0; i < 4; ++i) {
      const size_t off = kb + (size_t)(i * 32 + lane) * 4;
      f32x4 sa = *(const f32x4*)(s_att + off);
      f32x4 tp = *(const f32x4*)(t_spatial + off);
      accs += sa.x * sn[i*4+0] + sa.y * sn[i*4+1] + sa.z * sn[i*4+2] + sa.w * sn[i*4+3];
      acct += tp.x * tn[i*4+0] + tp.y * tn[i*4+1] + tp.z * tn[i*4+2] + tp.w * tn[i*4+3];
    }
    dps[k] = accs; dpt[k] = acct;
  }
#pragma unroll
  for (int off = 16; off > 0; off >>= 1) {
#pragma unroll
    for (int k = 0; k < TAUK; ++k) {
      dps[k] += __shfl_xor(dps[k], off, 32);
      dpt[k] += __shfl_xor(dpt[k], off, 32);
    }
  }

  const float scale = 0.0441941738241592f;   // 1/sqrt(512)
  float ws_[TAUK], wt_[TAUK];
  float ms = -3.4e38f, mt = -3.4e38f;
#pragma unroll
  for (int k = 0; k < TAUK; ++k) {
    dps[k] *= scale; dpt[k] *= scale;
    ms = fmaxf(ms, dps[k]); mt = fmaxf(mt, dpt[k]);
  }
  float es = 0.f, et = 0.f;
#pragma unroll
  for (int k = 0; k < TAUK; ++k) {
    ws_[k] = __expf(dps[k] - ms); es += ws_[k];
    wt_[k] = __expf(dpt[k] - mt); et += wt_[k];
  }
  const float ies = 1.f / es, iet = 1.f / et;
#pragma unroll
  for (int k = 0; k < TAUK; ++k) { ws_[k] *= ies; wt_[k] *= iet; }

  // trends + gated fusion, emit bf16 fusion states
#pragma unroll
  for (int i = 0; i < 4; ++i) {
    const size_t off = base + (size_t)(i * 32 + lane) * 4;
    float Tr0 = 0.f, Tr1 = 0.f, Tr2 = 0.f, Tr3 = 0.f;
    float Sr0 = 0.f, Sr1 = 0.f, Sr2 = 0.f, Sr3 = 0.f;
#pragma unroll
    for (int k = 0; k < TAUK; ++k) {
      const size_t kb = (size_t)k * NTOK * DIMK + off;
      f32x4 ta = *(const f32x4*)(t_att + kb);
      f32x4 sp = *(const f32x4*)(s_spatial + kb);
      Tr0 += ta.x * ws_[k]; Tr1 += ta.y * ws_[k]; Tr2 += ta.z * ws_[k]; Tr3 += ta.w * ws_[k];
      Sr0 += sp.x * wt_[k]; Sr1 += sp.y * wt_[k]; Sr2 += sp.z * wt_[k]; Sr3 += sp.w * wt_[k];
    }
    f32x4 Tv = *(const f32x4*)(T_t + off);
    f32x4 Sv = *(const f32x4*)(S_t + off);
    float Ttr[4] = {Tr0, Tr1, Tr2, Tr3};
    float Str[4] = {Sr0, Sr1, Sr2, Sr3};
    float Tvv[4] = {Tv.x, Tv.y, Tv.z, Tv.w};
    float Svv[4] = {Sv.x, Sv.y, Sv.z, Sv.w};
    float tf[4], sf[4];
#pragma unroll
    for (int c = 0; c < 4; ++c) {
      const float tg = sigmoidf_(tn[i * 4 + c]);
      const float sg = sigmoidf_(sn[i * 4 + c]);
      tf[c] = Tvv[c] * tg + (1.f - tg) * Ttr[c];
      sf[c] = Svv[c] * sg + (1.f - sg) * Str[c];
    }
    union { bf16x4 v; bf16x2 p[2]; } tb, sb;
    tb.p[0] = f2bf_pk(tf[0], tf[1]); tb.p[1] = f2bf_pk(tf[2], tf[3]);
    sb.p[0] = f2bf_pk(sf[0], sf[1]); sb.p[1] = f2bf_pk(sf[2], sf[3]);
    *(bf16x4*)(Tfus + off) = tb.v;
    *(bf16x4*)(Sfus + off) = sb.v;
  }
}

// ---------------------------------------------------------------------------
// P3: six-way concat GEMM + final cross-gating.
// grid=(NTOK/32, DIMK/128), block=256 (8 waves). Wave: 32 tokens x 16 cols,
// 12 WMMA accumulators; B fragments shared across the two M-subtiles.
// ---------------------------------------------------------------------------
__global__ void __launch_bounds__(256)
k_out_gemm(const __bf16* __restrict__ Tfus, const __bf16* __restrict__ Sfus,
           const __bf16* __restrict__ Wt, const __bf16* __restrict__ Ws,
           const float* __restrict__ bt, const float* __restrict__ bs,
           float* __restrict__ Tnew, float* __restrict__ Snew) {
  const int lane = threadIdx.x & 31;
  const int wave = threadIdx.x >> 5;
  const int lm = lane & 15;
  const int lh = lane >> 4;
  const int m0 = blockIdx.x * 32;
  const int j  = blockIdx.y * 128 + wave * 16 + lm;

  v8f a_tg[2] = {}, a_tt[2] = {}, a_ts[2] = {};
  v8f a_sg[2] = {}, a_st[2] = {}, a_ss[2] = {};

  const __bf16* At0 = Tfus + (size_t)(m0 + lm) * DIMK;
  const __bf16* At1 = Tfus + (size_t)(m0 + 16 + lm) * DIMK;
  const __bf16* As0 = Sfus + (size_t)(m0 + lm) * DIMK;
  const __bf16* As1 = Sfus + (size_t)(m0 + 16 + lm) * DIMK;
  const __bf16* Btg = Wt + (size_t)(j) * DIMK;
  const __bf16* Btt = Wt + (size_t)(512 + j) * DIMK;
  const __bf16* Bts = Wt + (size_t)(1024 + j) * DIMK;
  const __bf16* Bsg = Ws + (size_t)(j) * DIMK;
  const __bf16* Bst = Ws + (size_t)(512 + j) * DIMK;
  const __bf16* Bss = Ws + (size_t)(1024 + j) * DIMK;

  for (int k0 = 0; k0 < DIMK; k0 += 32) {
    const int ao = k0 + lh * 16;   // A chunk base (16 contiguous K)
    const int bo = k0 + lh * 16;   // B chunk base (16 contiguous K)

    BFrag aT0, aT1, aS0, aS1;
    aT0.q[0] = *(const u32x4*)(At0 + ao); aT0.q[1] = *(const u32x4*)(At0 + ao + 8);
    aT1.q[0] = *(const u32x4*)(At1 + ao); aT1.q[1] = *(const u32x4*)(At1 + ao + 8);
    aS0.q[0] = *(const u32x4*)(As0 + ao); aS0.q[1] = *(const u32x4*)(As0 + ao + 8);
    aS1.q[0] = *(const u32x4*)(As1 + ao); aS1.q[1] = *(const u32x4*)(As1 + ao + 8);

    BFrag btgf, bttf, btsf, bsgf, bstf, bssf;
    btgf.q[0] = *(const u32x4*)(Btg + bo); btgf.q[1] = *(const u32x4*)(Btg + bo + 8);
    bttf.q[0] = *(const u32x4*)(Btt + bo); bttf.q[1] = *(const u32x4*)(Btt + bo + 8);
    btsf.q[0] = *(const u32x4*)(Bts + bo); btsf.q[1] = *(const u32x4*)(Bts + bo + 8);
    bsgf.q[0] = *(const u32x4*)(Bsg + bo); bsgf.q[1] = *(const u32x4*)(Bsg + bo + 8);
    bstf.q[0] = *(const u32x4*)(Bst + bo); bstf.q[1] = *(const u32x4*)(Bst + bo + 8);
    bssf.q[0] = *(const u32x4*)(Bss + bo); bssf.q[1] = *(const u32x4*)(Bss + bo + 8);

    if (k0 + 32 < DIMK) {
      __builtin_prefetch(At0 + k0 + 32, 0, 1);
      __builtin_prefetch(As0 + k0 + 32, 0, 1);
    }

    a_tg[0] = WMMA_BF16(aT0.v, btgf.v, a_tg[0]);
    a_tg[1] = WMMA_BF16(aT1.v, btgf.v, a_tg[1]);
    a_tt[0] = WMMA_BF16(aT0.v, bttf.v, a_tt[0]);
    a_tt[1] = WMMA_BF16(aT1.v, bttf.v, a_tt[1]);
    a_ts[0] = WMMA_BF16(aT0.v, btsf.v, a_ts[0]);
    a_ts[1] = WMMA_BF16(aT1.v, btsf.v, a_ts[1]);
    a_sg[0] = WMMA_BF16(aS0.v, bsgf.v, a_sg[0]);
    a_sg[1] = WMMA_BF16(aS1.v, bsgf.v, a_sg[1]);
    a_st[0] = WMMA_BF16(aS0.v, bstf.v, a_st[0]);
    a_st[1] = WMMA_BF16(aS1.v, bstf.v, a_st[1]);
    a_ss[0] = WMMA_BF16(aS0.v, bssf.v, a_ss[0]);
    a_ss[1] = WMMA_BF16(aS1.v, bssf.v, a_ss[1]);
  }

  const float btg_b = bt[j], btt_b = bt[512 + j], bts_b = bt[1024 + j];
  const float bsg_b = bs[j], bst_b = bs[512 + j], bss_b = bs[1024 + j];

#pragma unroll
  for (int h = 0; h < 2; ++h) {
#pragma unroll
    for (int r = 0; r < 8; ++r) {
      const int row = m0 + h * 16 + lh * 8 + r;
      const float tg = sigmoidf_(a_tg[h][r] + btg_b);
      const float sg = sigmoidf_(a_sg[h][r] + bsg_b);
      const float Tv = tg * (a_tt[h][r] + btt_b) + (1.f - tg) * (a_st[h][r] + bst_b);
      const float Sv = sg * (a_ss[h][r] + bss_b) + (1.f - sg) * (a_ts[h][r] + bts_b);
      Tnew[(size_t)row * DIMK + j] = Tv;
      Snew[(size_t)row * DIMK + j] = Sv;
    }
  }
}

// ---------------------------------------------------------------------------
// Launch. Workspace layout (100 MiB total):
//   [Wtn_b 0.5MB][Wsn_b 0.5MB][Wt_b 1.5MB][Ws_b 1.5MB]
//   [tnext 32MB][snext 32MB][Tfus 16MB][Sfus 16MB]
// ---------------------------------------------------------------------------
extern "C" void kernel_launch(void* const* d_in, const int* in_sizes, int n_in,
                              void* d_out, int out_size, void* d_ws, size_t ws_size,
                              hipStream_t stream) {
  const float* T_t   = (const float*)d_in[0];
  const float* S_t   = (const float*)d_in[1];
  const float* t_att = (const float*)d_in[2];
  const float* s_att = (const float*)d_in[3];
  const float* t_sp  = (const float*)d_in[4];
  const float* s_sp  = (const float*)d_in[5];
  const float* W_t   = (const float*)d_in[6];
  const float* b_t   = (const float*)d_in[7];
  const float* W_tn  = (const float*)d_in[8];
  const float* b_tn  = (const float*)d_in[9];
  const float* W_s   = (const float*)d_in[10];
  const float* b_s   = (const float*)d_in[11];
  const float* W_sn  = (const float*)d_in[12];
  const float* b_sn  = (const float*)d_in[13];

  char* ws = (char*)d_ws;
  __bf16* Wtn_b = (__bf16*)ws;                 ws += (size_t)512 * 512 * 2;
  __bf16* Wsn_b = (__bf16*)ws;                 ws += (size_t)512 * 512 * 2;
  __bf16* Wt_b  = (__bf16*)ws;                 ws += (size_t)1536 * 512 * 2;
  __bf16* Ws_b  = (__bf16*)ws;                 ws += (size_t)1536 * 512 * 2;
  float*  tnext = (float*)ws;                  ws += (size_t)NTOK * DIMK * 4;
  float*  snext = (float*)ws;                  ws += (size_t)NTOK * DIMK * 4;
  __bf16* Tfus  = (__bf16*)ws;                 ws += (size_t)NTOK * DIMK * 2;
  __bf16* Sfus  = (__bf16*)ws;

  float* outT = (float*)d_out;
  float* outS = outT + (size_t)NTOK * DIMK;

  // P0: weight conversion
  {
    const int n4a = 512 * 512 / 4;    // 65536
    const int n4b = 1536 * 512 / 4;   // 196608
    k_cvt_bf16<<<(n4a + 255) / 256, 256, 0, stream>>>(W_tn, Wtn_b, n4a);
    k_cvt_bf16<<<(n4a + 255) / 256, 256, 0, stream>>>(W_sn, Wsn_b, n4a);
    k_cvt_bf16<<<(n4b + 255) / 256, 256, 0, stream>>>(W_t,  Wt_b,  n4b);
    k_cvt_bf16<<<(n4b + 255) / 256, 256, 0, stream>>>(W_s,  Ws_b,  n4b);
  }

  // P1: next-state GEMMs
  k_next_gemm<<<dim3(NTOK / 16, 2), 256, 0, stream>>>(
      T_t, S_t, Wtn_b, Wsn_b, b_tn, b_sn, tnext, snext);

  // P2: fusion streaming
  k_fuse<<<dim3(NTOK / 8), 256, 0, stream>>>(
      T_t, S_t, t_att, s_att, t_sp, s_sp, tnext, snext, Tfus, Sfus);

  // P3: concat GEMM + cross gating
  k_out_gemm<<<dim3(NTOK / 32, DIMK / 128), 256, 0, stream>>>(
      Tfus, Sfus, Wt_b, Ws_b, b_t, b_s, outT, outS);
}
